// ProfileHMM_8581344657585
// MI455X (gfx1250) — matrix operand order repaired and checked by reading, not compiled, not run on hardware
//
#include <hip/hip_runtime.h>

#define MP1      129           // M+1
#define DDIM     24
#define NSEQ_TOT 512
#define LSEQ     256
#define KREAL    258           // 2*(M+1)
#define KP       288           // padded K: 18 tiles of 16, 9 chunks of 32
#define KTILES   (KP / 16)     // 18
#define KCHUNKS  (KP / 32)     // 9
#define NSEQ_BLK 16
#define NTHREADS 288           // 9 waves x 2 tiles = 18 tiles, perfectly balanced
#define NWAVES   (NTHREADS / 32)
#define NULLV    (-1e30f)

typedef __attribute__((ext_vector_type(16))) _Float16 v16h;
typedef __attribute__((ext_vector_type(8)))  _Float16 v8h;
typedef __attribute__((ext_vector_type(8)))  float    v8f;

// ---------------------------------------------------------------------------
// Transition-logit helpers (pyro.mue profile arrangement)
// rn/un layout: [i*6 + j*2 + c] for (M+1,3,2)
// ---------------------------------------------------------------------------
__device__ __forceinline__ float z0f(int i, int j, int mp,
                                     const float* rn, const float* un, const float* S) {
  if (mp == i) return rn[i*6 + j*2 + 0] + un[i*6 + j*2 + 0];
  if (mp >  i) return rn[i*6 + j*2 + 0] + un[i*6 + j*2 + 1] +
                      (S[mp-1] - S[i]) + rn[mp*6 + 4] + un[mp*6 + 4];
  return NULLV;
}
__device__ __forceinline__ float z1f(int i, int j, int mp,
                                     const float* rn, const float* un, const float* S) {
  if (mp == i) return rn[i*6 + j*2 + 1];
  if (mp >  i) return rn[i*6 + j*2 + 0] + un[i*6 + j*2 + 1] +
                      (S[mp-1] - S[i]) + rn[mp*6 + 5];
  return NULLV;
}
__device__ __forceinline__ float trans_logit(int jrow, int c,
                                             const float* rn, const float* un, const float* S) {
  if (jrow < MP1) {                    // from match state m=jrow  => i=m+1, j=0
    int i = jrow + 1;
    if (i > MP1 - 1) return NULLV;     // virtual NULL slab row
    return (c < MP1) ? z0f(i, 0, c, rn, un, S) : z1f(i, 0, c - MP1, rn, un, S);
  } else {                             // from insert state m=jrow-129 => i=m, j=1
    int i = jrow - MP1;
    return (c < MP1) ? z0f(i, 1, c, rn, un, S) : z1f(i, 1, c - MP1, rn, un, S);
  }
}

// ---------------------------------------------------------------------------
// Kernel 1: build exp(T)^T (f16, padded KPxKP), exp(init), exp(obs) tables
// ---------------------------------------------------------------------------
__global__ __launch_bounds__(288)
void build_model(const float* __restrict__ anc, const float* __restrict__ insq,
                 const float* __restrict__ ins, const float* __restrict__ del,
                 float* __restrict__ EobsG,     // [DDIM+1][KP]
                 float* __restrict__ rn, float* __restrict__ un,
                 float* __restrict__ S,         // [MP1]
                 float* __restrict__ EinitG,    // [KP]
                 _Float16* __restrict__ TexpT)  // [KP][KP]  (dest-major)
{
  const int tid = threadIdx.x;        // 288 threads

  // -- observation log-softmax -> exp(obs) table, transposed [d][k] --
  for (int k = tid; k < KREAL; k += 288) {
    const float* src = (k < MP1) ? (anc + k * DDIM) : (insq + (k - MP1) * DDIM);
    float mx = -1e30f;
    for (int d = 0; d < DDIM; ++d) mx = fmaxf(mx, src[d]);
    float s = 0.f;
    for (int d = 0; d < DDIM; ++d) s += expf(src[d] - mx);
    float lse = mx + logf(s);
    for (int d = 0; d < DDIM; ++d) EobsG[d * KP + k] = expf(src[d] - lse);
  }
  for (int k = tid; k < KP; k += 288) {
    if (k >= KREAL)
      for (int d = 0; d < DDIM; ++d) EobsG[d * KP + k] = 0.f;
    EobsG[DDIM * KP + k] = (k < KREAL) ? 1.f : 0.f;   // "no emission" row
  }

  // -- indel log-softmax over last axis (size 2) --
  for (int idx = tid; idx < MP1 * 3; idx += 288) {
    int b = idx * 2;
    {
      float a = ins[b], c = ins[b + 1];
      float mx = fmaxf(a, c);
      float lse = mx + logf(expf(a - mx) + expf(c - mx));
      rn[b] = a - lse; rn[b + 1] = c - lse;
    }
    {
      float a = del[b], c = del[b + 1];
      float mx = fmaxf(a, c);
      float lse = mx + logf(expf(a - mx) + expf(c - mx));
      un[b] = a - lse; un[b + 1] = c - lse;
    }
  }
  __syncthreads();

  // -- cumulative chain S[m] = cumsum(r[m,2,0] + u[m,2,1]) --
  if (tid == 0) {
    float acc = 0.f;
    for (int m = 0; m < MP1; ++m) { acc += rn[m*6 + 4] + un[m*6 + 5]; S[m] = acc; }
  }
  __syncthreads();

  // -- zero padded exp(T)^T then fill real entries --
  for (int i = tid; i < KP * KP; i += 288) TexpT[i] = (_Float16)0.f;
  __syncthreads();
  for (int e = tid; e < KREAL * KREAL; e += 288) {
    int jrow = e / KREAL, c = e % KREAL;
    float tl = trans_logit(jrow, c, rn, un, S);
    float v  = (tl < -60.f) ? 0.f : expf(tl);
    TexpT[(size_t)c * KP + jrow] = (_Float16)v;       // transposed: [dest][src]
  }
  for (int k = tid; k < KP; k += 288) {
    float v = 0.f;
    if (k < KREAL) {
      float il = (k < MP1) ? z0f(0, 0, k, rn, un, S) : z1f(0, 0, k - MP1, rn, un, S);
      v = (il < -60.f) ? 0.f : expf(il);
    }
    EinitG[k] = v;
  }
}

// ---------------------------------------------------------------------------
// Kernel 2: one-hot data -> token ids (24 == masked / no emission)
// ---------------------------------------------------------------------------
__global__ __launch_bounds__(256)
void token_kernel(const float* __restrict__ data, int* __restrict__ tok) {
  int idx = blockIdx.x * blockDim.x + threadIdx.x;
  if (idx >= NSEQ_TOT * LSEQ) return;
  const float* row = data + (size_t)idx * DDIM;
  int tk = DDIM;
  for (int d = 0; d < DDIM; ++d)
    if (row[d] > 0.5f) { tk = d; break; }
  tok[idx] = tk;
}

// ---------------------------------------------------------------------------
// Kernel 3: scaled-linear HMM forward, WMMA f16xf16->f32, exp(T)^T in LDS
// 9 waves x 2 column tiles each = 18 tiles (KP=288): no per-wave imbalance,
// no EXEC-predicated WMMA.
// ---------------------------------------------------------------------------
__global__ __launch_bounds__(NTHREADS)
void hmm_forward(const _Float16* __restrict__ TexpTG,
                 const float* __restrict__ EobsG,
                 const float* __restrict__ EinitG,
                 const int* __restrict__ tokG,
                 float* __restrict__ out)
{
  extern __shared__ ulonglong4 smemv[];            // 32B-aligned dynamic LDS
  char* smem = (char*)smemv;
  _Float16* Ts    = (_Float16*)smem;                                    // KP*KP f16
  float*    Eob   = (float*)(smem + (size_t)KP * KP * 2);               // 25*KP f32
  _Float16* pBuf0 = (_Float16*)((char*)Eob + (DDIM + 1) * KP * 4);      // 16*KP f16
  _Float16* pBuf1 = pBuf0 + NSEQ_BLK * KP;
  float* rowpart  = (float*)(pBuf1 + NSEQ_BLK * KP);                    // 16*16 f32
  float* rowmax   = rowpart + 256;                                      // 16
  float* cacc     = rowmax + 16;                                        // 16
  int*   tokS     = (int*)(cacc + 16);                                  // 16

  const int tid  = threadIdx.x;
  const int lane = tid & 31, wave = tid >> 5;      // wave in [0, 9)
  const int half = lane >> 4, lh = lane & 15;
  const int nbase = blockIdx.x * NSEQ_BLK;

  // stage exp(T)^T and exp(obs) into LDS (read once from HBM, reused 255x)
  {
    const uint4* src = (const uint4*)TexpTG;
    uint4* dst = (uint4*)Ts;
    for (int i = tid; i < (KP * KP * 2) / 16; i += NTHREADS) dst[i] = src[i];
    for (int i = tid; i < (DDIM + 1) * KP; i += NTHREADS) Eob[i] = EobsG[i];
  }
  __syncthreads();

  // ---- t = 0: p0 = exp(init) * exp(obs[tok0]); rescale to row max ----
  if (tid < NSEQ_BLK) {
    tokS[tid] = tokG[(size_t)(nbase + tid) * LSEQ];
    cacc[tid] = 0.f;
  }
  __syncthreads();
  if (tid < 256) {
    int n = tid >> 4, kl = tid & 15;
    int tk = tokS[n];
    float mx = 0.f;
    for (int k = kl; k < KP; k += 16)
      mx = fmaxf(mx, EinitG[k] * Eob[tk * KP + k]);
    rowpart[n * 16 + kl] = mx;
  }
  __syncthreads();
  if (tid < NSEQ_BLK) {
    float m = 0.f;
    for (int w = 0; w < 16; ++w) m = fmaxf(m, rowpart[tid * 16 + w]);
    m = fmaxf(m, 1e-30f);
    rowmax[tid] = m;
    cacc[tid] += logf(m);
  }
  __syncthreads();
  if (tid < 256) {
    int n = tid >> 4, kl = tid & 15;
    int tk = tokS[n];
    float inv = 1.f / rowmax[n];
    for (int k = kl; k < KP; k += 16)
      pBuf0[n * KP + k] = (_Float16)(EinitG[k] * Eob[tk * KP + k] * inv);
  }
  __syncthreads();

  // ---- recursion: p <- rescale( (p x expT) * expObs[tok_t] ) ----
  _Float16* pC = pBuf0;
  _Float16* pN = pBuf1;
  const int t0 = wave * 16;                 // first column tile of this wave
  const int t1 = (wave + NWAVES) * 16;      // second column tile

  for (int t = 1; t < LSEQ; ++t) {
    if (tid < NSEQ_BLK) tokS[tid] = tokG[(size_t)(nbase + tid) * LSEQ + t];
    __syncthreads();

    v8f acc0 = {}; v8f acc1 = {};
#pragma unroll
    for (int kc = 0; kc < KCHUNKS; ++kc) {
      // A fragment: row = lane&15, K = 16*half + e  (row-major p)
      const _Float16* pa = pC + lh * KP + kc * 32 + half * 16;
      v8h alo = *(const v8h*)pa;
      v8h ahi = *(const v8h*)(pa + 8);
      v16h A = __builtin_shufflevector(alo, ahi, 0,1,2,3,4,5,6,7,8,9,10,11,12,13,14,15);
      // B fragments: col = lane&15, K = 16*half + e  (dest-major expT rows)
      v16h B0 = *(const v16h*)(Ts + (size_t)(t0 + lh) * KP + kc * 32 + half * 16);
      v16h B1 = *(const v16h*)(Ts + (size_t)(t1 + lh) * KP + kc * 32 + half * 16);
      acc0 = __builtin_amdgcn_wmma_f32_16x16x32_f16(false, A, false, B0, (short)0, acc0, false, false);
      acc1 = __builtin_amdgcn_wmma_f32_16x16x32_f16(false, A, false, B1, (short)0, acc1, false, false);
    }

    // emission multiply + per-row max (C/D layout: row = i + 8*half, col = lh)
    int tk[8];
#pragma unroll
    for (int i = 0; i < 8; ++i) tk[i] = tokS[i + 8 * half];
    float v0[8], v1[8], lm[8];
#pragma unroll
    for (int i = 0; i < 8; ++i) {
      v0[i] = acc0[i] * Eob[tk[i] * KP + t0 + lh];
      v1[i] = acc1[i] * Eob[tk[i] * KP + t1 + lh];
      lm[i] = fmaxf(v0[i], v1[i]);
    }
#pragma unroll
    for (int off = 1; off < 16; off <<= 1) {
#pragma unroll
      for (int i = 0; i < 8; ++i) lm[i] = fmaxf(lm[i], __shfl_xor(lm[i], off, 32));
    }
    if (lh == 0) {
#pragma unroll
      for (int i = 0; i < 8; ++i) rowpart[wave * 16 + (i + 8 * half)] = lm[i];
    }
    __syncthreads();
    if (tid < NSEQ_BLK) {
      float m = 0.f;
      for (int w = 0; w < NWAVES; ++w) m = fmaxf(m, rowpart[w * 16 + tid]);
      m = fmaxf(m, 1e-30f);
      rowmax[tid] = m;
      cacc[tid] += logf(m);
    }
    __syncthreads();
#pragma unroll
    for (int i = 0; i < 8; ++i) {
      int row = i + 8 * half;
      float inv = 1.f / rowmax[row];
      pN[row * KP + t0 + lh] = (_Float16)(v0[i] * inv);
      pN[row * KP + t1 + lh] = (_Float16)(v1[i] * inv);
    }
    __syncthreads();
    _Float16* tmp = pC; pC = pN; pN = tmp;
  }

  // ---- out[n] = c[n] + log(sum_k p[n,k]) ----
  if (tid < NSEQ_BLK) {
    float s = 0.f;
    for (int k = 0; k < KP; ++k) s += (float)pC[tid * KP + k];
    out[nbase + tid] = cacc[tid] + logf(s);
  }
}

// ---------------------------------------------------------------------------
extern "C" void kernel_launch(void* const* d_in, const int* in_sizes, int n_in,
                              void* d_out, int out_size, void* d_ws, size_t ws_size,
                              hipStream_t stream) {
  const float* anc  = (const float*)d_in[0];
  const float* insq = (const float*)d_in[1];
  const float* ins  = (const float*)d_in[2];
  const float* del  = (const float*)d_in[3];
  const float* data = (const float*)d_in[4];
  float* out = (float*)d_out;

  char* w = (char*)d_ws;
  auto alloc = [&](size_t bytes) -> char* {
    char* p = w; w += (bytes + 255) & ~(size_t)255; return p;
  };
  float*    EobsG  = (float*)   alloc((DDIM + 1) * KP * sizeof(float));
  float*    rn     = (float*)   alloc(MP1 * 6 * sizeof(float));
  float*    un     = (float*)   alloc(MP1 * 6 * sizeof(float));
  float*    S      = (float*)   alloc(MP1 * sizeof(float));
  float*    EinitG = (float*)   alloc(KP * sizeof(float));
  _Float16* TexpTG = (_Float16*)alloc((size_t)KP * KP * sizeof(_Float16));
  int*      tokG   = (int*)     alloc((size_t)NSEQ_TOT * LSEQ * sizeof(int));

  build_model<<<1, 288, 0, stream>>>(anc, insq, ins, del, EobsG, rn, un, S, EinitG, TexpTG);
  token_kernel<<<(NSEQ_TOT * LSEQ + 255) / 256, 256, 0, stream>>>(data, tokG);

  size_t smemBytes = (size_t)KP * KP * 2            // exp(T)^T f16
                   + (size_t)(DDIM + 1) * KP * 4    // exp(obs)
                   + 2 * (size_t)NSEQ_BLK * KP * 2  // p double-buffer f16
                   + 256 * 4 + 16 * 4 + 16 * 4 + 16 * 4;
  hmm_forward<<<NSEQ_TOT / NSEQ_BLK, NTHREADS, smemBytes, stream>>>(TexpTG, EobsG, EinitG, tokG, out);
}
